// MultiHeadSelfAttention_48481590837427
// MI455X (gfx1250) — compile-verified, matrix operand order checked by reading, not run on hardware
//
#include <hip/hip_runtime.h>
#include <hip/hip_bf16.h>
#include <math.h>

// ---------------------------------------------------------------------------
// MI455X (gfx1250) multi-head self-attention, bf16 WMMA everywhere.
//   x:[2,2048,1024] f32 -> out:[2,2048,1024] f32
//   GEMMs: C = A(row-major, MxK) * B(row-major, NxK)^T + bias via
//   v_wmma_f32_16x16x32_bf16, with a statically software-pipelined
//   global_load_async_to_lds_b128 double buffer (ASYNCcnt pipeline).
// ---------------------------------------------------------------------------

typedef __attribute__((ext_vector_type(16))) __bf16 v16bf;
typedef __attribute__((ext_vector_type(8)))  __bf16 v8bf;
typedef __attribute__((ext_vector_type(4)))  __bf16 v4bf;
typedef __attribute__((ext_vector_type(8)))  float  v8f;

#define SEQ   2048
#define HID   1024
#define NH    16
#define HS    64
#define MTOT  4096   // 2 * 2048 tokens

__device__ __forceinline__ __bf16 f2bf(float f) {
  unsigned u = __float_as_uint(f);
  u += 0x7FFFu + ((u >> 16) & 1u);          // round-to-nearest-even
  unsigned short h = (unsigned short)(u >> 16);
  return __builtin_bit_cast(__bf16, h);
}

__device__ __forceinline__ v8f wmma_bf16(v16bf a, v16bf b, v8f c) {
  return __builtin_amdgcn_wmma_f32_16x16x32_bf16(false, a, false, b,
                                                 (short)0, c, false, false);
}

// Async copy 16B global -> LDS (per lane). lds_off = raw LDS byte offset
// (low 32 bits of the flat shared-memory address, per the LDS aperture rule).
__device__ __forceinline__ void async_b128(unsigned lds_off, const void* g) {
  asm volatile("global_load_async_to_lds_b128 %0, %1, off"
               :: "v"(lds_off), "v"((unsigned long long)(size_t)g)
               : "memory");
}
__device__ __forceinline__ void wait_async0() {
  asm volatile("s_wait_asynccnt 0" ::: "memory");
}

// 16x32 bf16 fragment (A layout; identical pattern serves as B for C=A*B^T):
// lane l: row = l&15, chunk offset koff = (l<16 ? 0 : 8); elems [koff..koff+7]
// in regs 0..3 and [koff+16..koff+23] in regs 4..7.
__device__ __forceinline__ v16bf load_frag16(const __bf16* base, int ld, int lane) {
  const int m    = lane & 15;
  const int koff = (lane >> 4) << 3;
  const __bf16* p = base + (size_t)m * ld + koff;
  v8bf lo = *(const v8bf*)(p);
  v8bf hi = *(const v8bf*)(p + 16);
  v16bf r;
#pragma unroll
  for (int i = 0; i < 8; ++i) { r[i] = lo[i]; r[i + 8] = hi[i]; }
  return r;
}

// ---------------------------------------------------------------------------
// f32 -> bf16 conversion, 4 elems/thread
// ---------------------------------------------------------------------------
__global__ __launch_bounds__(256) void cvt_kernel(const float* __restrict__ s,
                                                  __bf16* __restrict__ d, int n) {
  int i = (blockIdx.x * 256 + threadIdx.x) * 4;
  if (i < n) {
    float4 f = *(const float4*)(s + i);
    v4bf o; o[0] = f2bf(f.x); o[1] = f2bf(f.y); o[2] = f2bf(f.z); o[3] = f2bf(f.w);
    *(v4bf*)(d + i) = o;
  }
}

// ---------------------------------------------------------------------------
// GEMM: C[M=4096, N=1024] = A * B^T + bias,  K = 1024, bf16 in / f32 acc.
// Block: 256 threads = 8 waves, tile 128(M) x 128(N); wave = 32x64 (2x4 WMMA).
// 32 K-slabs of 32; slab s lives in LDS buffer (s&1). Static software
// pipeline: prologue fills buf0; steady-state loop handles two slabs per
// iteration with compile-time buffer indices (no branches, immediate LDS
// offsets); epilogue drains the last two slabs.
// mode 0: outf[m*1024+n] = c                    (final projection, f32)
// mode 1: outb per-head [b][h][q][d]            (Q, K)
// mode 2: outb per-head transposed [b][h][d][q] (V^T for PV B-fragments)
// ---------------------------------------------------------------------------
__global__ __launch_bounds__(256) void gemm_kernel(
    const __bf16* __restrict__ A, const __bf16* __restrict__ B,
    const float* __restrict__ bias, float* __restrict__ outf,
    __bf16* __restrict__ outb, int mode) {
  constexpr int Kd = 1024;
  __shared__ __align__(16) __bf16 As[2][128 * 32];
  __shared__ __align__(16) __bf16 Bs[2][128 * 32];

  const int tid  = threadIdx.x;
  const int lane = tid & 31;
  const int wid  = tid >> 5;
  const int wm   = wid >> 1;            // 0..3  -> M offset wm*32
  const int wn   = wid & 1;             // 0..1  -> N offset wn*64
  const int m0   = blockIdx.x * 128;
  const int n0   = blockIdx.y * 128;

  // 128x32 tile staging: each of 256 threads owns 16 contiguous elems (2x16B)
  const int trow = tid >> 1;            // 0..127
  const int tcol = (tid & 1) << 4;      // 0 / 16
  const __bf16* ag = A + (size_t)(m0 + trow) * Kd + tcol;  // next slab to issue
  const __bf16* bg = B + (size_t)(n0 + trow) * Kd + tcol;
  unsigned aoff[2], boff[2];
  aoff[0] = (unsigned)(size_t)&As[0][trow * 32 + tcol];
  aoff[1] = (unsigned)(size_t)&As[1][trow * 32 + tcol];
  boff[0] = (unsigned)(size_t)&Bs[0][trow * 32 + tcol];
  boff[1] = (unsigned)(size_t)&Bs[1][trow * 32 + tcol];

  v8f acc[2][4] = {};

  // per-buffer fragment compute (compile-time buffer index)
  auto compute = [&](int buf) {
    const __bf16* Ab = As[buf];
    const __bf16* Bb = Bs[buf];
    v16bf af[2], bfm[4];
#pragma unroll
    for (int mt = 0; mt < 2; ++mt)
      af[mt] = load_frag16(&Ab[(wm * 32 + mt * 16) * 32], 32, lane);
#pragma unroll
    for (int nt = 0; nt < 4; ++nt)
      bfm[nt] = load_frag16(&Bb[(wn * 64 + nt * 16) * 32], 32, lane);
#pragma unroll
    for (int mt = 0; mt < 2; ++mt)
#pragma unroll
      for (int nt = 0; nt < 4; ++nt)
        acc[mt][nt] = wmma_bf16(af[mt], bfm[nt], acc[mt][nt]);
  };
  auto issue = [&](int buf, const __bf16* a, const __bf16* b) {
    async_b128(aoff[buf],      a);
    async_b128(aoff[buf] + 16, a + 8);
    async_b128(boff[buf],      b);
    async_b128(boff[buf] + 16, b + 8);
  };

  // prologue: slab 0 -> buf0
  issue(0, ag, bg);
  ag += 32; bg += 32;
  wait_async0();
  __syncthreads();

  // steady state: slabs 1..30 issued, slabs 0..29 computed (15 pair-iters)
  for (int p = 0; p < 15; ++p) {
    issue(1, ag, bg);                   // slab 2p+1 -> buf1
    compute(0);                         // slab 2p
    wait_async0();
    __syncthreads();
    issue(0, ag + 32, bg + 32);         // slab 2p+2 -> buf0
    compute(1);                         // slab 2p+1
    ag += 64; bg += 64;
    wait_async0();
    __syncthreads();
  }
  // epilogue: slab 31 -> buf1, compute slabs 30 and 31
  issue(1, ag, bg);
  compute(0);
  wait_async0();
  __syncthreads();
  compute(1);

  const int rbase = (lane >> 4) << 3;   // C layout: row = rbase + r, col = lane&15
  const int ncol  = lane & 15;
#pragma unroll
  for (int mt = 0; mt < 2; ++mt)
#pragma unroll
    for (int nt = 0; nt < 4; ++nt) {
      const int n  = n0 + wn * 64 + nt * 16 + ncol;
      const float bv = bias[n];
#pragma unroll
      for (int r = 0; r < 8; ++r) {
        const int m = m0 + wm * 32 + mt * 16 + rbase + r;
        const float v = acc[mt][nt][r] + bv;
        if (mode == 0) {
          outf[(size_t)m * HID + n] = v;
        } else {
          const int b = m >> 11, q = m & 2047, h = n >> 6, d = n & 63;
          if (mode == 1)
            outb[(((size_t)(b * NH + h)) * SEQ + q) * HS + d] = f2bf(v);
          else
            outb[(((size_t)(b * NH + h)) * HS + d) * SEQ + q] = f2bf(v);
        }
      }
    }
}

// ---------------------------------------------------------------------------
// Causal flash attention. Grid: (32 q-blocks of 64, 32 batch*head).
// Block: 128 threads = 4 waves; each wave owns 16 query rows.
// Per 32-key block: 4 WMMA for S = Q*K^T, 4 WMMA for O += P*V.
// P goes C-layout -> A-layout through a private per-wave LDS slab
// (same-wave DS ordering, so no block barrier -> causal divergence is safe).
// ---------------------------------------------------------------------------
__global__ __launch_bounds__(128) void attn_kernel(
    const __bf16* __restrict__ Qb, const __bf16* __restrict__ Kb,
    const __bf16* __restrict__ Vt, __bf16* __restrict__ att) {
  __shared__ __align__(16) __bf16 psm_all[4][16 * 32];

  const int lane = threadIdx.x & 31;
  const int wid  = threadIdx.x >> 5;
  const int bh   = blockIdx.y;                 // b*16 + h
  const int b    = bh >> 4, h = bh & 15;
  const int q0   = blockIdx.x * 64 + wid * 16; // first query row of this wave
  __bf16* psm = psm_all[wid];

  const __bf16* Q  = Qb + (size_t)bh * SEQ * HS;
  const __bf16* Kp = Kb + (size_t)bh * SEQ * HS;
  const __bf16* Vp = Vt + (size_t)bh * HS * SEQ;

  // resident Q fragments (16 rows x 64 head dims = 2 K-steps)
  v16bf qf0 = load_frag16(Q + (size_t)q0 * HS,      HS, lane);
  v16bf qf1 = load_frag16(Q + (size_t)q0 * HS + 32, HS, lane);

  v8f o[4] = {};
  float mi[8], li[8];
#pragma unroll
  for (int r = 0; r < 8; ++r) { mi[r] = -INFINITY; li[r] = 0.0f; }

  const int rbase = (lane >> 4) << 3;
  const int ncol  = lane & 15;
  const float scale = 0.125f;                  // 1/sqrt(64)

  const int jmax = (q0 + 15) >> 5;             // causal key-block limit
  for (int j = 0; j <= jmax; ++j) {
    const int kb = j * 32;
    if (j < jmax) {                            // warm caches for next slab
      __builtin_prefetch(Kp + (size_t)(kb + 32 + (lane & 15)) * HS, 0, 3);
      __builtin_prefetch(Vp + (size_t)(lane & 15) * SEQ + kb + 32, 0, 3);
    }
    v8f s0 = {}, s1 = {};
    {
      v16bf kf;
      kf = load_frag16(Kp + (size_t)(kb +  0) * HS,      HS, lane);
      s0 = wmma_bf16(qf0, kf, s0);
      kf = load_frag16(Kp + (size_t)(kb +  0) * HS + 32, HS, lane);
      s0 = wmma_bf16(qf1, kf, s0);
      kf = load_frag16(Kp + (size_t)(kb + 16) * HS,      HS, lane);
      s1 = wmma_bf16(qf0, kf, s1);
      kf = load_frag16(Kp + (size_t)(kb + 16) * HS + 32, HS, lane);
      s1 = wmma_bf16(qf1, kf, s1);
    }

    // online softmax over this 32-key slab + stash P(bf16) into LDS
#pragma unroll
    for (int r = 0; r < 8; ++r) {
      const int q = q0 + rbase + r;
      float a0 = s0[r] * scale; if (kb + ncol      > q) a0 = -INFINITY;
      float a1 = s1[r] * scale; if (kb + 16 + ncol > q) a1 = -INFINITY;
      float mx = fmaxf(a0, a1);
      mx = fmaxf(mx, __shfl_xor(mx, 1, 32));
      mx = fmaxf(mx, __shfl_xor(mx, 2, 32));
      mx = fmaxf(mx, __shfl_xor(mx, 4, 32));
      mx = fmaxf(mx, __shfl_xor(mx, 8, 32));
      const float mnew  = fmaxf(mi[r], mx);
      const float alpha = __expf(mi[r] - mnew);
      const float p0 = __expf(a0 - mnew);
      const float p1 = __expf(a1 - mnew);
      float rs = p0 + p1;
      rs += __shfl_xor(rs, 1, 32);
      rs += __shfl_xor(rs, 2, 32);
      rs += __shfl_xor(rs, 4, 32);
      rs += __shfl_xor(rs, 8, 32);
      mi[r] = mnew;
      li[r] = li[r] * alpha + rs;
#pragma unroll
      for (int dt = 0; dt < 4; ++dt) o[dt][r] *= alpha;
      const int m = rbase + r;
      psm[m * 32 + ncol]      = f2bf(p0);
      psm[m * 32 + 16 + ncol] = f2bf(p1);
    }
    // same-wave DS ops are in-order; just drain the store counter
    asm volatile("s_wait_dscnt 0" ::: "memory");

    v16bf pf = load_frag16(psm, 32, lane);     // P as 16x32 A-fragment
#pragma unroll
    for (int dt = 0; dt < 4; ++dt) {
      v16bf vf = load_frag16(Vp + (size_t)(dt * 16) * SEQ + kb, SEQ, lane);
      o[dt] = wmma_bf16(pf, vf, o[dt]);
    }
  }

  // normalize + write att[b][q][h*64+d] (bf16, A-operand of final projection)
#pragma unroll
  for (int r = 0; r < 8; ++r) {
    const float inv = 1.0f / li[r];
    const int q = q0 + rbase + r;
    __bf16* dst = att + ((size_t)b * SEQ + q) * HID + h * HS;
#pragma unroll
    for (int dt = 0; dt < 4; ++dt) dst[dt * 16 + ncol] = f2bf(o[dt][r] * inv);
  }
}

// ---------------------------------------------------------------------------
// Host orchestration
// ---------------------------------------------------------------------------
extern "C" void kernel_launch(void* const* d_in, const int* in_sizes, int n_in,
                              void* d_out, int out_size, void* d_ws, size_t ws_size,
                              hipStream_t stream) {
  (void)in_sizes; (void)n_in; (void)out_size; (void)ws_size;
  const float* x  = (const float*)d_in[0];
  const float* Wq = (const float*)d_in[1];
  const float* bq = (const float*)d_in[2];
  const float* Wk = (const float*)d_in[3];
  const float* bk = (const float*)d_in[4];
  const float* Wv = (const float*)d_in[5];
  const float* bv = (const float*)d_in[6];
  const float* Wm = (const float*)d_in[7];
  const float* bm = (const float*)d_in[8];
  float* out = (float*)d_out;

  __bf16* ws = (__bf16*)d_ws;                 // 25,165,824 bf16 = 48 MB
  __bf16* Xb   = ws;                          // 4096x1024
  __bf16* Wqb  = Xb   + (size_t)MTOT * HID;
  __bf16* Wkb  = Wqb  + (size_t)HID * HID;
  __bf16* Wvb  = Wkb  + (size_t)HID * HID;
  __bf16* Wmb  = Wvb  + (size_t)HID * HID;
  __bf16* Qb   = Wmb  + (size_t)HID * HID;    // [2][16][2048][64]
  __bf16* Kb   = Qb   + (size_t)MTOT * HID;
  __bf16* Vt   = Kb   + (size_t)MTOT * HID;   // [2][16][64][2048]
  __bf16* attb = Vt   + (size_t)MTOT * HID;   // [4096][1024]

  cvt_kernel<<<(MTOT * HID) / 1024, 256, 0, stream>>>(x,  Xb,  MTOT * HID);
  cvt_kernel<<<(HID * HID) / 1024,  256, 0, stream>>>(Wq, Wqb, HID * HID);
  cvt_kernel<<<(HID * HID) / 1024,  256, 0, stream>>>(Wk, Wkb, HID * HID);
  cvt_kernel<<<(HID * HID) / 1024,  256, 0, stream>>>(Wv, Wvb, HID * HID);
  cvt_kernel<<<(HID * HID) / 1024,  256, 0, stream>>>(Wm, Wmb, HID * HID);

  dim3 ggrid(MTOT / 128, HID / 128);
  gemm_kernel<<<ggrid, 256, 0, stream>>>(Xb, Wqb, bq, nullptr, Qb, 1);
  gemm_kernel<<<ggrid, 256, 0, stream>>>(Xb, Wkb, bk, nullptr, Kb, 1);
  gemm_kernel<<<ggrid, 256, 0, stream>>>(Xb, Wvb, bv, nullptr, Vt, 2);

  attn_kernel<<<dim3(SEQ / 64, 2 * NH), 128, 0, stream>>>(Qb, Kb, Vt, attb);

  gemm_kernel<<<ggrid, 256, 0, stream>>>(attb, Wmb, bm, out, nullptr, 0);
}
